// GGNNMessagePassing_22325240004849
// MI455X (gfx1250) — matrix operand order repaired and checked by reading, not compiled, not run on hardware
//
#include <hip/hip_runtime.h>

// ---------------------------------------------------------------------------
// GGNN message passing step for MI455X (gfx1250, wave32, WMMA bf16)
// ---------------------------------------------------------------------------

typedef __bf16 bf16_t;
typedef bf16_t v16bf __attribute__((ext_vector_type(16)));
typedef bf16_t v8bf  __attribute__((ext_vector_type(8)));
typedef float  v8f   __attribute__((ext_vector_type(8)));

#define HID 128
#define NT  4

// round-to-nearest-even f32 -> bf16
__device__ __forceinline__ unsigned short f2bf(float f) {
    unsigned u = __float_as_uint(f);
    u += 0x7FFFu + ((u >> 16) & 1u);
    return (unsigned short)(u >> 16);
}

// B-matrix fragment (K=32 x N=16, bf16). Lane layout used here (contiguous k, same as A):
// lanes 0-15 hold K=0..15 of column N=lane; lanes 16-31 hold K=16..31 of
// N=lane-16.  Column = edge/node; K runs along the hidden dim of one row ->
// one contiguous 32B load per lane.
__device__ __forceinline__ v16bf load_b_frag(const unsigned short* X, long row,
                                             int kc, int half) {
    return *reinterpret_cast<const v16bf*>(X + row * HID + kc * 32 + half * 16);
}

// A-matrix fragment (M=16 x K=32, bf16). Lane layout:
// lanes 0-15: M=lane, K = 0..15; lanes 16-31: M=lane-16, K = 16..31.
// -> two contiguous 16B loads per lane from the bf16 weight row.
__device__ __forceinline__ v16bf load_a_frag(const unsigned short* W, int row,
                                             int kc, int half) {
    const unsigned short* p = W + row * HID + kc * 32 + half * 16;
    union { v16bf v; v8bf h[2]; } u;
    u.h[0] = *reinterpret_cast<const v8bf*>(p);
    u.h[1] = *reinterpret_cast<const v8bf*>(p + 8);
    return u.v;
}

__device__ __forceinline__ v8f wmma_bf16(v16bf a, v16bf b, v8f c) {
    return __builtin_amdgcn_wmma_f32_16x16x32_bf16(
        /*neg_a=*/false, a, /*neg_b=*/false, b,
        /*c_mod=*/(short)0, c, /*reuse_a=*/false, /*reuse_b=*/false);
}

// ---------------------------------------------------------------------------
// Prep kernels
// ---------------------------------------------------------------------------
__global__ void k_zero_f32(float* p, long n) {
    long i = (long)blockIdx.x * blockDim.x + threadIdx.x;
    if (i < n) p[i] = 0.0f;
}

__global__ void k_cvt_bf16(const float* __restrict__ src,
                           unsigned short* __restrict__ dst, long n) {
    long i = (long)blockIdx.x * blockDim.x + threadIdx.x;
    if (i < n) dst[i] = f2bf(src[i]);
}

// ---------------------------------------------------------------------------
// Edge-message kernel: one wave handles 32 edges (two 16-column groups that
// share every A fragment).  For each type t we run the unmasked WMMA chain
// into a fresh accumulator and then predicated-add it into the final
// accumulator per lane (D columns are per-lane, so acc-side masking is exact
// and avoids materializing masked B fragments).
// __launch_bounds__(256, 1): allow the full 256-VGPR allocation so the
// 8 pipelined accumulator chains + resident B fragments do not spill.
// ---------------------------------------------------------------------------
__global__ void __launch_bounds__(256, 1)
k_edge_msg(const unsigned short* __restrict__ nsb,   // [N,128] bf16
           const unsigned short* __restrict__ Wb,    // [4,128,128] bf16
           const float* __restrict__ edge_b,         // [4,128] f32
           const int* __restrict__ edge_index,       // [2,E]
           const int* __restrict__ edge_type,        // [E]
           float* __restrict__ agg,                  // [N,128] f32
           int E) {
    int wid  = (blockIdx.x * blockDim.x + threadIdx.x) >> 5;  // wave id = tile id
    int lane = threadIdx.x & 31;
    int tiles = (E + 31) >> 5;
    if (wid >= tiles) return;  // uniform per wave -> EXEC stays all-ones below

    int col  = lane & 15;   // D-matrix column (edge within group)
    int half = lane >> 4;   // K/M half selector per ISA layouts

    long e0 = (long)wid * 32 + col;       // group 0 edge
    long e1 = e0 + 16;                    // group 1 edge
    bool valid0 = (e0 < (long)E);
    bool valid1 = (e1 < (long)E);
    int  type0 = valid0 ? edge_type[e0] : -1;
    int  type1 = valid1 ? edge_type[e1] : -1;
    long src0  = valid0 ? (long)edge_index[e0] : 0;
    long src1  = valid1 ? (long)edge_index[e1] : 0;
    long dst0  = valid0 ? (long)edge_index[(long)E + e0] : 0;
    long dst1  = valid1 ? (long)edge_index[(long)E + e1] : 0;

    // Gathered x_src B-fragments, both groups (4 K-chunks cover K=128)
    v16bf xb0[4], xb1[4];
#pragma unroll
    for (int kc = 0; kc < 4; ++kc) {
        xb0[kc] = load_b_frag(nsb, src0, kc, half);
        xb1[kc] = load_b_frag(nsb, src1, kc, half);
    }

#pragma unroll 1
    for (int jc = 0; jc < 8; ++jc) {                    // 8 x 16 output dims
        v8f facc0 = {}, facc1 = {};
#pragma unroll
        for (int t = 0; t < NT; ++t) {
            const unsigned short* Wt = Wb + t * HID * HID;
            v16bf a[4];
#pragma unroll
            for (int kc = 0; kc < 4; ++kc)
                a[kc] = load_a_frag(Wt, jc * 16 + col, kc, half);
            v8f acc0 = {}, acc1 = {};
#pragma unroll
            for (int kc = 0; kc < 4; ++kc) {
                acc0 = wmma_bf16(a[kc], xb0[kc], acc0);
                acc1 = wmma_bf16(a[kc], xb1[kc], acc1);
            }
            bool m0 = (type0 == t), m1 = (type1 == t);
#pragma unroll
            for (int r = 0; r < 8; ++r) {
                facc0[r] += m0 ? acc0[r] : 0.0f;        // per-lane column mask
                facc1[r] += m1 ? acc1[r] : 0.0f;
            }
        }
        if (valid0) {
            float*       aggrow = agg + dst0 * HID;
            const float* brow   = edge_b + (long)type0 * HID;
#pragma unroll
            for (int r = 0; r < 8; ++r) {
                int j = jc * 16 + r + 8 * half;         // D: M = r + 8*half
                atomicAdd(&aggrow[j], facc0[r] + brow[j]);
            }
        }
        if (valid1) {
            float*       aggrow = agg + dst1 * HID;
            const float* brow   = edge_b + (long)type1 * HID;
#pragma unroll
            for (int r = 0; r < 8; ++r) {
                int j = jc * 16 + r + 8 * half;
                atomicAdd(&aggrow[j], facc1[r] + brow[j]);
            }
        }
    }
}

// ---------------------------------------------------------------------------
// GRU kernel: one wave handles 16 nodes.  gi = agg @ w_ih^T + b_ih,
// gh = h @ w_hh^T + b_hh, gate math, output.
// ---------------------------------------------------------------------------
__global__ void __launch_bounds__(256, 1)
k_gru(const float* __restrict__ node_states,   // [N,128] f32
      const unsigned short* __restrict__ nsb,  // [N,128] bf16
      const float* __restrict__ agg,           // [N,128] f32
      const unsigned short* __restrict__ wihb, // [384,128] bf16
      const unsigned short* __restrict__ whhb, // [384,128] bf16
      const float* __restrict__ b_ih,          // [384]
      const float* __restrict__ b_hh,          // [384]
      float* __restrict__ out,                 // [N,128] f32
      int N) {
    int wid  = (blockIdx.x * blockDim.x + threadIdx.x) >> 5;
    int lane = threadIdx.x & 31;
    int tiles = (N + 15) >> 4;
    if (wid >= tiles) return;

    int col  = lane & 15;
    int half = lane >> 4;
    long node  = (long)wid * 16 + col;
    bool valid = (node < (long)N);
    long nodec = valid ? node : 0;

    // B-fragments: h (bf16 precomputed) and agg (f32 -> bf16 on the fly)
    v16bf hf[4], af[4];
#pragma unroll
    for (int kc = 0; kc < 4; ++kc) hf[kc] = load_b_frag(nsb, nodec, kc, half);
#pragma unroll
    for (int kc = 0; kc < 4; ++kc) {
        const float4* p = reinterpret_cast<const float4*>(
            agg + nodec * HID + kc * 32 + half * 16);
        union { v16bf v; unsigned short s[16]; } u;
#pragma unroll
        for (int q = 0; q < 4; ++q) {
            float4 f = p[q];
            u.s[q * 4 + 0] = f2bf(f.x);
            u.s[q * 4 + 1] = f2bf(f.y);
            u.s[q * 4 + 2] = f2bf(f.z);
            u.s[q * 4 + 3] = f2bf(f.w);
        }
        af[kc] = u.v;
    }

#pragma unroll 1
    for (int jc = 0; jc < 8; ++jc) {
        v8f ir = {}, iz = {}, in_ = {}, hr = {}, hz = {}, hn = {};
#pragma unroll
        for (int kc = 0; kc < 4; ++kc) {
            int rowA = jc * 16 + col;
            ir  = wmma_bf16(load_a_frag(wihb, 0 * HID + rowA, kc, half), af[kc], ir);
            iz  = wmma_bf16(load_a_frag(wihb, 1 * HID + rowA, kc, half), af[kc], iz);
            in_ = wmma_bf16(load_a_frag(wihb, 2 * HID + rowA, kc, half), af[kc], in_);
            hr  = wmma_bf16(load_a_frag(whhb, 0 * HID + rowA, kc, half), hf[kc], hr);
            hz  = wmma_bf16(load_a_frag(whhb, 1 * HID + rowA, kc, half), hf[kc], hz);
            hn  = wmma_bf16(load_a_frag(whhb, 2 * HID + rowA, kc, half), hf[kc], hn);
        }
        if (valid) {
#pragma unroll
            for (int r = 0; r < 8; ++r) {
                int j = jc * 16 + r + 8 * half;         // D: M = r + 8*half
                float vir = ir[r]  + b_ih[j];
                float viz = iz[r]  + b_ih[HID + j];
                float vin = in_[r] + b_ih[2 * HID + j];
                float vhr = hr[r]  + b_hh[j];
                float vhz = hz[r]  + b_hh[HID + j];
                float vhn = hn[r]  + b_hh[2 * HID + j];
                float rg = 1.0f / (1.0f + __expf(-(vir + vhr)));
                float zg = 1.0f / (1.0f + __expf(-(viz + vhz)));
                float ng = tanhf(vin + rg * vhn);
                float h  = node_states[node * HID + j];
                out[node * HID + j] = (1.0f - zg) * ng + zg * h;
            }
        }
    }
}

// ---------------------------------------------------------------------------
// Launch
// ---------------------------------------------------------------------------
extern "C" void kernel_launch(void* const* d_in, const int* in_sizes, int n_in,
                              void* d_out, int out_size, void* d_ws, size_t ws_size,
                              hipStream_t stream) {
    const float* node_states = (const float*)d_in[0];
    const int*   edge_index  = (const int*)d_in[1];
    const int*   edge_type   = (const int*)d_in[2];
    const float* edge_W      = (const float*)d_in[3];
    const float* edge_b      = (const float*)d_in[4];
    const float* w_ih        = (const float*)d_in[5];
    const float* w_hh        = (const float*)d_in[6];
    const float* b_ih        = (const float*)d_in[7];
    const float* b_hh        = (const float*)d_in[8];
    float* out = (float*)d_out;

    const int N = in_sizes[0] / HID;   // 50000
    const int E = in_sizes[2];         // 625000

    // Workspace layout (all sections 256B aligned)
    char* ws = (char*)d_ws;
    float*          agg  = (float*)ws;                              // N*128 f32
    size_t off = (size_t)N * HID * sizeof(float);
    off = (off + 255) & ~(size_t)255;
    unsigned short* nsb  = (unsigned short*)(ws + off);             // N*128 bf16
    off += (size_t)N * HID * sizeof(unsigned short);
    off = (off + 255) & ~(size_t)255;
    unsigned short* Wb   = (unsigned short*)(ws + off);             // 4*128*128 bf16
    off += (size_t)NT * HID * HID * sizeof(unsigned short);
    off = (off + 255) & ~(size_t)255;
    unsigned short* wihb = (unsigned short*)(ws + off);             // 384*128 bf16
    off += (size_t)3 * HID * HID * sizeof(unsigned short);
    off = (off + 255) & ~(size_t)255;
    unsigned short* whhb = (unsigned short*)(ws + off);             // 384*128 bf16

    const int B = 256;
    long n;

    // 1) zero agg
    n = (long)N * HID;
    k_zero_f32<<<(int)((n + B - 1) / B), B, 0, stream>>>(agg, n);

    // 2) bf16 conversions
    n = (long)N * HID;
    k_cvt_bf16<<<(int)((n + B - 1) / B), B, 0, stream>>>(node_states, nsb, n);
    n = (long)NT * HID * HID;
    k_cvt_bf16<<<(int)((n + B - 1) / B), B, 0, stream>>>(edge_W, Wb, n);
    n = (long)3 * HID * HID;
    k_cvt_bf16<<<(int)((n + B - 1) / B), B, 0, stream>>>(w_ih, wihb, n);
    k_cvt_bf16<<<(int)((n + B - 1) / B), B, 0, stream>>>(w_hh, whhb, n);

    // 3) per-edge-type messages + scatter (one wave per 32 edges)
    int etiles = (E + 31) >> 5;
    int waves_per_blk = B / 32;
    int egrid = (etiles + waves_per_blk - 1) / waves_per_blk;
    k_edge_msg<<<egrid, B, 0, stream>>>(nsb, Wb, edge_b, edge_index, edge_type,
                                        agg, E);

    // 4) GRU cell (one wave per 16 nodes)
    int ntiles = (N + 15) >> 4;
    int ngrid = (ntiles + waves_per_blk - 1) / waves_per_blk;
    k_gru<<<ngrid, B, 0, stream>>>(node_states, nsb, agg, wihb, whhb,
                                   b_ih, b_hh, out, N);
}
